// Conv_surface_37873021616799
// MI455X (gfx1250) — matrix operand order repaired and checked by reading, not compiled, not run on hardware
//
#include <hip/hip_runtime.h>
#include <hip/hip_bf16.h>
#include <math.h>

typedef __attribute__((ext_vector_type(16))) _Float16 v16h;
typedef __attribute__((ext_vector_type(8)))  _Float16 v8h;
typedef __attribute__((ext_vector_type(8)))  float    v8f;

#define BSZ   4
#define NV    4096
#define NNB   32
#define CIN   128
#define COUT  256
#define MROWS (BSZ * NV * NNB)   // 524288
#define VN    (NV * NNB)         // 131072 = 2^17

// ---------------------------------------------------------------------------
// prep: convert W (256x128 f32) -> f16, zero the per-channel stat accumulators
// ---------------------------------------------------------------------------
__global__ __launch_bounds__(256) void prep_kernel(const float* __restrict__ W,
                                                   _Float16* __restrict__ Wh,
                                                   float* __restrict__ sums,
                                                   float* __restrict__ sumsq) {
    const int i = blockIdx.x * 256 + threadIdx.x;
    if (i < COUT * CIN) Wh[i] = (_Float16)W[i];
    if (i < COUT) { sums[i] = 0.0f; sumsq[i] = 0.0f; }
}

// ---------------------------------------------------------------------------
// finalize BN: scale = gamma*rsqrt(var+eps), shift = beta - mean*scale
// ---------------------------------------------------------------------------
__global__ __launch_bounds__(256) void finalize_kernel(const float* __restrict__ sums,
                                                       const float* __restrict__ sumsq,
                                                       const float* __restrict__ gamma,
                                                       const float* __restrict__ beta,
                                                       float* __restrict__ scale,
                                                       float* __restrict__ shift) {
    const int c = threadIdx.x;
    const float invM = 1.0f / (float)MROWS;
    const float mean = sums[c] * invM;
    const float var  = fmaxf(sumsq[c] * invM - mean * mean, 0.0f);
    const float sc   = gamma[c] * rsqrtf(var + 1e-5f);
    scale[c] = sc;
    shift[c] = beta[c] - mean * sc;
}

// ---------------------------------------------------------------------------
// Main fused pass. Each block owns 128 rows (= 4 vertices x 32 neighbors).
// STATS=true : accumulate per-channel sum / sumsq of z (f16 WMMA GEMM).
// STATS=false: recompute z, apply BN+relu, multiply theta, max over neighbors.
// ---------------------------------------------------------------------------
template <bool STATS>
__global__ __launch_bounds__(256) void conv_pass_kernel(
    const int*   __restrict__ nbr,    // (bs,v,n) neighbor vertex ids
    const float* __restrict__ verts,  // (bs,v,3)
    const float* __restrict__ feat,   // (bs,v,127)
    const float* __restrict__ dirs,   // (4,256)
    const _Float16* __restrict__ Wh,  // (256,128) f16
    float* __restrict__ gsum,         // [256]   (STATS)
    float* __restrict__ gsumsq,       // [256]   (STATS)
    const float* __restrict__ scale,  // [256]   (!STATS)
    const float* __restrict__ shift,  // [256]   (!STATS)
    float* __restrict__ out)          // (bs,v,256)
{
    __shared__ __align__(16) _Float16 fsS[128][136];  // gathered features, f16
    __shared__ float    dwS[128][4];                  // direction weights
    __shared__ float    statS[2 * COUT];              // block sum / sumsq
    __shared__ unsigned outS[4][COUT];                // per-vertex running max (bits)

    const int  tid     = threadIdx.x;
    const long rowBase = (long)blockIdx.x * 128;

    for (int i = tid; i < 2 * COUT; i += 256) statS[i] = 0.0f;
    for (int i = tid; i < 4 * COUT; i += 256) ((unsigned*)outS)[i] = 0u;

    // ---- stage: 2 threads per row (halves of the 128-wide K dim) ----
    {
        const int  rl = tid >> 1;         // local row 0..127
        const int  hf = tid & 1;
        const long r  = rowBase + rl;     // flat (b,v,n) row
        const int  b  = (int)(r >> 17);
        const int  rem = (int)(r & (VN - 1));
        const int  vi  = rem >> 5;
        const int  nb  = nbr[r];          // gathered vertex id
        const float* frow = feat + ((long)b * NV + nb) * (CIN - 1);

        float ss = 0.0f;
        const int k0 = hf * 64;
        const int k1 = hf ? (CIN - 1) : 64;
        for (int k = k0; k < k1; ++k) {
            const float x = frow[k];
            ss += x * x;
            fsS[rl][k] = (_Float16)x;
        }
        ss += __shfl_xor(ss, 1, 32);                       // pair-combine sumsq
        if (hf) fsS[rl][CIN - 1] = (_Float16)sqrtf(ss);    // distance channel
        if (!STATS && hf == 0) {
            const float* vb = verts + (long)b * NV * 3;
            const float dx = vb[nb * 3 + 0] - vb[vi * 3 + 0];
            const float dy = vb[nb * 3 + 1] - vb[vi * 3 + 1];
            const float dz = vb[nb * 3 + 2] - vb[vi * 3 + 2];
            const float inv = 1.0f / fmaxf(sqrtf(dx * dx + dy * dy + dz * dz), 1e-12f);
            dwS[rl][0] = (dx * inv + 1.0f) * 0.5f;
            dwS[rl][1] = (dy * inv + 1.0f) * 0.5f;
            dwS[rl][2] = (dz * inv + 1.0f) * 0.5f;
        }
    }
    __syncthreads();

    // ---- compute: wave w owns rows [16w, 16w+16), loops 16 column tiles ----
    const int wave = tid >> 5;
    const int lane = tid & 31;
    const int l    = lane & 15;
    const int hi   = lane >> 4;
    const int rowT = wave * 16;

    // A fragments (contiguous k, matches B): j=0..15 -> k=s*32+hi*16+j
    //   => two contiguous 16B LDS reads each.
    v16h afrag[4];
    const _Float16* arow = &fsS[rowT + l][0];
#pragma unroll
    for (int s = 0; s < 4; ++s) {
        const int kb = s * 32 + hi * 16;
        *(v8h*)&afrag[s]       = *(const v8h*)(arow + kb);
        *((v8h*)&afrag[s] + 1) = *(const v8h*)(arow + kb + 8);
    }

    for (int ct = 0; ct < COUT / 16; ++ct) {
        const int colBase = ct * 16;
        v8f acc = {};
#pragma unroll
        for (int s = 0; s < 4; ++s) {
            // B (K x N) lane layout: n = l, k = s*32 + hi*16 + j  -> 32 contiguous bytes
            const _Float16* bp = Wh + (long)(colBase + l) * CIN + s * 32 + hi * 16;
            const v16h bfrag = *(const v16h*)bp;
            acc = __builtin_amdgcn_wmma_f32_16x16x32_f16(
                false, afrag[s], false, bfrag, (short)0, acc, false, false);
        }

        if (STATS) {
            float ls = 0.0f, lss = 0.0f;
#pragma unroll
            for (int p = 0; p < 8; ++p) { const float z = acc[p]; ls += z; lss += z * z; }
            atomicAdd(&statS[colBase + l], ls);
            atomicAdd(&statS[COUT + colBase + l], lss);
        } else {
            const int   c  = colBase + l;
            const float sc = scale[c];
            const float sh = shift[c];
            const float d0 = dirs[c];
            const float s0 = dirs[COUT + c] - d0;
            const float s1 = dirs[2 * COUT + c] - d0;
            const float s2 = dirs[3 * COUT + c] - d0;
            float m = 0.0f;                       // act >= 0, so 0 is identity
#pragma unroll
            for (int p = 0; p < 8; ++p) {
                const int rl = rowT + hi * 8 + p; // C layout: hi picks rows 0-7 / 8-15
                float th = d0 + s0 * dwS[rl][0] + s1 * dwS[rl][1] + s2 * dwS[rl][2];
                th = fmaxf(th, 0.0f);
                const float a = fmaxf(acc[p] * sc + sh, 0.0f) * th;
                m = fmaxf(m, a);
            }
            m = fmaxf(m, __shfl_xor(m, 16, 32));  // combine row halves of column
            atomicMax(&outS[wave >> 1][c], __float_as_uint(m));  // monotone for >=0
        }
    }
    __syncthreads();

    // ---- flush ----
    if (STATS) {
        atomicAdd(&gsum[tid], statS[tid]);
        atomicAdd(&gsumsq[tid], statS[COUT + tid]);
    } else {
        const int b   = (int)(rowBase >> 17);
        const int vi0 = (int)((rowBase & (VN - 1)) >> 5);
        for (int i = tid; i < 4 * COUT; i += 256) {
            const int vtx = i >> 8;
            const int c   = i & (COUT - 1);
            out[((long)b * NV + vi0 + vtx) * COUT + c] = __uint_as_float(outS[vtx][c]);
        }
    }
}

// ---------------------------------------------------------------------------
extern "C" void kernel_launch(void* const* d_in, const int* in_sizes, int n_in,
                              void* d_out, int out_size, void* d_ws, size_t ws_size,
                              hipStream_t stream) {
    const int*   nbr   = (const int*)d_in[0];
    const float* verts = (const float*)d_in[1];
    const float* feat  = (const float*)d_in[2];
    const float* dirs  = (const float*)d_in[3];
    const float* W     = (const float*)d_in[4];
    /* d_in[5] = b: cancels under BatchNorm, unused */
    const float* gamma = (const float*)d_in[6];
    const float* beta  = (const float*)d_in[7];
    float* out = (float*)d_out;

    // workspace layout
    _Float16* Wh    = (_Float16*)d_ws;                      // 64 KB
    float*    sums  = (float*)((char*)d_ws + COUT * CIN * sizeof(_Float16));
    float*    sumsq = sums + COUT;
    float*    scale = sums + 2 * COUT;
    float*    shift = sums + 3 * COUT;

    const int nblocks = MROWS / 128;  // 4096

    prep_kernel<<<(COUT * CIN + 255) / 256, 256, 0, stream>>>(W, Wh, sums, sumsq);
    conv_pass_kernel<true><<<nblocks, 256, 0, stream>>>(
        nbr, verts, feat, dirs, Wh, sums, sumsq, nullptr, nullptr, nullptr);
    finalize_kernel<<<1, 256, 0, stream>>>(sums, sumsq, gamma, beta, scale, shift);
    conv_pass_kernel<false><<<nblocks, 256, 0, stream>>>(
        nbr, verts, feat, dirs, Wh, nullptr, nullptr, scale, shift, out);
}
